// Measurement_68307159875839
// MI455X (gfx1250) — matrix operand order repaired and checked by reading, not compiled, not run on hardware
//
#include <hip/hip_runtime.h>

typedef __attribute__((ext_vector_type(16))) _Float16     v16h;
typedef __attribute__((ext_vector_type(8)))  float        v8f;
typedef __attribute__((ext_vector_type(4)))  unsigned int u32x4;
typedef __attribute__((ext_vector_type(8)))  int          i32x8;
typedef __attribute__((ext_vector_type(4)))  int          i32x4;
typedef __attribute__((ext_vector_type(4)))  float        f32x4;

#define H   768
#define KC  32              // k per WMMA chunk
#define NI  (H / KC)        // 24 k-chunks
#define IB  128             // i-block per workgroup (8 tiles of 16)
#define JB  128             // j-block per workgroup (8 waves x 16 rows)

// ---------------------------------------------------------------------------
// Prep: convert operator fp32 -> f16 once (2.36 MB read, 1.18 MB write).
// ---------------------------------------------------------------------------
__global__ __launch_bounds__(256)
void op_cvt_kernel(const float* __restrict__ Op, _Float16* __restrict__ OpH)
{
    const int t = (blockIdx.x * 256 + threadIdx.x) * 4;
    const f32x4 v = *(const f32x4*)(Op + t);
    _Float16* d = OpH + t;
    d[0] = (_Float16)v.x; d[1] = (_Float16)v.y;
    d[2] = (_Float16)v.z; d[3] = (_Float16)v.w;
}

// TDM: DMA one 32(k) x 128(i) f16 tile of OpH into LDS at ldsOff.
__device__ __forceinline__
void tdm_issue(const _Float16* gsrc, unsigned ldsOff)
{
    const unsigned long long ga = (unsigned long long)(size_t)gsrc;
    u32x4 g0;
    g0[0] = 1u;                                  // count=1 (valid), user mode
    g0[1] = ldsOff;                              // lds_addr (bytes)
    g0[2] = (unsigned)(ga & 0xFFFFFFFFu);        // global_addr[31:0]
    g0[3] = (unsigned)((ga >> 32) & 0x01FFFFFFu) // global_addr[56:32]
          | (2u << 30);                          // type=2 ("image")
    i32x8 g1;
    g1[0] = 0x00010000;          // wg_mask=0, data_size=1 (2 bytes)
    g1[1] = (int)(768u << 16);   // tensor_dim0[15:0]=768
    g1[2] = (int)(768u << 16);   // tensor_dim0[31:16]=0 ; tensor_dim1[15:0]=768
    g1[3] = (int)(32u  << 16);   // tensor_dim1[31:16]=0 ; tile_dim0=32
    g1[4] = 128;                 // tile_dim1=128, tile_dim2=0
    g1[5] = 768;                 // tensor_dim0_stride[31:0]
    g1[6] = 0;                   // stride0[47:32]=0, stride1[15:0]=0
    g1[7] = 0;
    const i32x4 gz4 = (i32x4){0, 0, 0, 0};       // 2D tensor: groups 2/3 unused
    const i32x8 gz8 = (i32x8){0, 0, 0, 0, 0, 0, 0, 0};
    __builtin_amdgcn_tensor_load_to_lds(g0, g1, gz4, gz4, gz8, 0);
}

// ---------------------------------------------------------------------------
// diag[b,i] = sum_j op[i,j] * C2[j,i],  C2 = X[b] @ op^T  (M=j, N=i, K=k)
// B operand (op^T chunk, f16) staged to LDS by the Tensor Data Mover,
// double-buffered so the DMA overlaps the WMMA work of the previous chunk.
// ---------------------------------------------------------------------------
__global__ __launch_bounds__(256)
void diag_gemm_kernel(const float* __restrict__ X,
                      const float* __restrict__ Op,
                      const _Float16* __restrict__ OpH,
                      float* __restrict__ diag)
{
    __shared__ __align__(64) _Float16 ldsB[2][IB][KC];   // 2 x 8 KB ping-pong
    __shared__ float ldsDiag[IB];

    const int tid  = threadIdx.x;
    const int lane = tid & 31;
    const int wave = tid >> 5;      // 0..7
    const int l16  = lane & 15;
    const int lh   = lane >> 4;     // 0 or 1 (lane half)

    const int ib0 = blockIdx.x * IB;     // i block base
    const int jb0 = blockIdx.y * JB;     // j block base
    const int b   = blockIdx.z;          // batch

    const float* Xb = X + (size_t)b * H * H;
    const int jw = jb0 + wave * 16;      // this wave's j-tile base

    const _Float16* opSrc = OpH + (size_t)ib0 * H;           // row ib0, col 0
    const unsigned  ldsOff0 = (unsigned)(size_t)&ldsB[0][0][0]; // flat[31:0]==LDS offset
    const unsigned  bufBytes = IB * KC * sizeof(_Float16);

    if (tid < IB) ldsDiag[tid] = 0.0f;

    v8f acc[8];
#pragma unroll
    for (int t = 0; t < 8; ++t) acc[t] = (v8f){};

    // A-operand per-lane k offsets (contiguous, matching B from LDS):
    // lanes 0-15: k = 0..15 ; lanes 16-31: k = 16..31
    const int c0 = lh ? 16 : 0;

    // Prologue: start DMA of chunk 0 into buffer 0.
    if (wave == 0) tdm_issue(opSrc, ldsOff0);

    for (int ic = 0; ic < NI; ++ic) {
        const int p  = ic & 1;
        const int kc = ic * KC;

        __syncthreads();   // WAR: buffer (1-p) reads from iteration ic-1 complete

        if (wave == 0 && ic + 1 < NI)      // start DMA of next chunk into other buffer
            tdm_issue(opSrc + (size_t)(ic + 1) * KC, ldsOff0 + (1 - p) * bufBytes);

        // In-order TDM completion: <=1 outstanding  =>  chunk ic has landed.
        if (ic + 1 < NI) __builtin_amdgcn_s_wait_tensorcnt(1);
        else             __builtin_amdgcn_s_wait_tensorcnt(0);
        __syncthreads();   // publish buffer p to all waves

        // A chunk: X rows (jw + l16), two contiguous 8-float runs -> v16h
        v16h a;
        {
            const float* ar0 = Xb + (size_t)(jw + l16) * H + kc + c0;
            const float* ar1 = ar0 + 8;
            if (ic + 1 < NI)   // prefetch next A chunk (global_prefetch_b8)
                __builtin_prefetch(ar0 + KC, 0, 1);
#pragma unroll
            for (int e = 0; e < 8; ++e) {
                a[e]     = (_Float16)ar0[e];
                a[8 + e] = (_Float16)ar1[e];
            }
        }

        // 8 i-tiles, 2-deep rotating B pipeline: load tile t+1 while WMMA t runs
        v16h bcur = *(const v16h*)&ldsB[p][l16][lh * 16];
#pragma unroll
        for (int t = 0; t < 8; ++t) {
            v16h bnxt = bcur;
            if (t + 1 < 8)
                bnxt = *(const v16h*)&ldsB[p][(t + 1) * 16 + l16][lh * 16];
            acc[t] = __builtin_amdgcn_wmma_f32_16x16x32_f16(
                         false, a, false, bcur, (short)0, acc[t], false, false);
            bcur = bnxt;
        }
    }

    // Epilogue: lane L, tile t holds C2[jw + r + 8*lh][ib0 + t*16 + l16] in acc[t][r]
#pragma unroll
    for (int t = 0; t < 8; ++t) {
        const int iloc = t * 16 + l16;
        const float* oprow = Op + (size_t)(ib0 + iloc) * H + jw + lh * 8;  // 8 consecutive
        float s = 0.0f;
#pragma unroll
        for (int r = 0; r < 8; ++r) s += acc[t][r] * oprow[r];
        __hip_atomic_fetch_add(&ldsDiag[iloc], s,
                               __ATOMIC_RELAXED, __HIP_MEMORY_SCOPE_WORKGROUP);
    }
    __syncthreads();
    if (tid < IB) {
        __hip_atomic_fetch_add(diag + (size_t)b * H + ib0 + tid, ldsDiag[tid],
                               __ATOMIC_RELAXED, __HIP_MEMORY_SCOPE_AGENT);
    }
}

// ---------------------------------------------------------------------------
// Row softmax: one workgroup per batch, 768 values, 3 per thread.
// ---------------------------------------------------------------------------
__global__ __launch_bounds__(256)
void softmax_kernel(const float* __restrict__ diag, float* __restrict__ out)
{
    __shared__ float red[256];
    const int b = blockIdx.x, tid = threadIdx.x;
    const float* row = diag + (size_t)b * H;

    float v0 = row[tid], v1 = row[tid + 256], v2 = row[tid + 512];
    float m = fmaxf(v0, fmaxf(v1, v2));
    red[tid] = m;
    __syncthreads();
    for (int s = 128; s > 0; s >>= 1) {
        if (tid < s) red[tid] = fmaxf(red[tid], red[tid + s]);
        __syncthreads();
    }
    m = red[0];
    __syncthreads();

    float e0 = __expf(v0 - m), e1 = __expf(v1 - m), e2 = __expf(v2 - m);
    red[tid] = e0 + e1 + e2;
    __syncthreads();
    for (int s = 128; s > 0; s >>= 1) {
        if (tid < s) red[tid] += red[tid + s];
        __syncthreads();
    }
    const float inv = 1.0f / red[0];

    float* orow = out + (size_t)b * H;
    orow[tid]       = e0 * inv;
    orow[tid + 256] = e1 * inv;
    orow[tid + 512] = e2 * inv;
}

extern "C" void kernel_launch(void* const* d_in, const int* in_sizes, int n_in,
                              void* d_out, int out_size, void* d_ws, size_t ws_size,
                              hipStream_t stream)
{
    const float* X  = (const float*)d_in[0];   // [128,768,768] fp32
    const float* Op = (const float*)d_in[1];   // [768,768] fp32
    float* out  = (float*)d_out;               // [128,768] fp32

    float*    diag = (float*)d_ws;                              // 128*768 fp32 (384 KB)
    _Float16* OpH  = (_Float16*)((char*)d_ws + 128 * H * 4);    // 768*768 f16 (1.125 MB)

    (void)hipMemsetAsync(diag, 0, (size_t)128 * H * sizeof(float), stream);

    op_cvt_kernel<<<H * H / (256 * 4), 256, 0, stream>>>(Op, OpH);

    dim3 grid(H / IB, H / JB, 128);            // (6, 6, 128) = 4608 workgroups
    diag_gemm_kernel<<<grid, 256, 0, stream>>>(X, Op, OpH, diag);
    softmax_kernel<<<128, 256, 0, stream>>>(diag, out);
}
